// GRRN_41807211660023
// MI455X (gfx1250) — matrix operand order repaired and checked by reading, not compiled, not run on hardware
//
#include <hip/hip_runtime.h>
#include <hip/hip_bf16.h>
#include <math.h>

typedef __attribute__((ext_vector_type(16))) _Float16 v16h;
typedef __attribute__((ext_vector_type(8)))  _Float16 v8h;
typedef __attribute__((ext_vector_type(8)))  float    v8f;

#define BB 2
#define NN 384
#define DD 128
#define TT 4
#define ROWS (BB * NN)          // 768 rows (batch*seq)
#define NTILES (DD / 16)        // 8 col tiles
#define FRAG   512              // halves per 32x16 fragment (32 lanes * 16)

// ---------------- WMMA fragment loaders --------------------------------------
// A: 16x32 (MxK) from row-major f16. Lanes 0-15: M=lane, K {k0..k0+15};
// lanes 16-31: same M, K {k0+16..k0+31} (matches swizzled B).
__device__ __forceinline__ v16h load_a_f16(const _Float16* __restrict__ A, int lda,
                                           int row0, int k0) {
  int lane = threadIdx.x & 31;
  const _Float16* p = A + (size_t)(row0 + (lane & 15)) * lda + k0 + ((lane >> 4) << 4);
  v16h a;
#pragma unroll
  for (int e = 0; e < 8; ++e) { a[e] = p[e]; a[8 + e] = p[8 + e]; }
  return a;
}
// B: pre-swizzled fragment array, one contiguous 32B chunk per lane.
__device__ __forceinline__ v16h load_b_sw(const _Float16* __restrict__ Bsw,
                                          int frag /* = c*NTILES + t */) {
  return *(const v16h*)(Bsw + ((size_t)frag * 32 + (threadIdx.x & 31)) * 16);
}

#define WMMA_F16(a, b, c) \
  __builtin_amdgcn_wmma_f32_16x16x32_f16(false, (a), false, (b), (short)0, (c), false, false)

__device__ __forceinline__ float sigmoidf_(float x) { return 1.0f / (1.0f + expf(-x)); }

// fragment-flat index for f32 gating state: [rowTile][wave][lane][r]
__device__ __forceinline__ size_t ffidx(int rowTile, int wave, int lane, int r) {
  return (((size_t)rowTile * 8 + wave) * 32 + lane) * 8 + r;
}

// ---------------- setup kernels ----------------------------------------------
// Swizzle a KxDD f32 weight into B-fragment layout: dst[((c*8+t)*32+lane)*16+e]
// holds W[(32c + ((lane>>4)<<4) + e) * DD + 16t + (lane&15)].
__global__ void k_swz_b(const float* __restrict__ W, _Float16* __restrict__ dst, int total) {
  int idx = blockIdx.x * blockDim.x + threadIdx.x;
  if (idx >= total) return;
  int e    = idx & 15;
  int lane = (idx >> 4) & 31;
  int t    = (idx >> 9) & 7;
  int c    = idx >> 12;
  int k = c * 32 + ((lane >> 4) << 4) + e;
  int n = t * 16 + (lane & 15);
  dst[idx] = (_Float16)W[(size_t)k * DD + n];
}

// prop init: row-major f16 shadow + fragment-flat f32 state
__global__ void k_init_prop(const float* __restrict__ x, float* __restrict__ pf,
                            _Float16* __restrict__ propH, int n) {
  int i = blockIdx.x * blockDim.x + threadIdx.x;
  if (i >= n) return;
  float v = x[i];
  propH[i] = (_Float16)v;
  int row = i / DD, col = i % DD;
  int lane = (col & 15) + ((row & 15) >= 8 ? 16 : 0);
  pf[ffidx(row >> 4, col >> 4, lane, row & 7)] = v;
}

// wrow = Wa1[:D]@Wa2, wcol = Wa1[D:]@Wa2, c = ba1.Wa2 + ba2
__global__ void k_prep_attn(const float* __restrict__ Wa1, const float* __restrict__ ba1,
                            const float* __restrict__ Wa2, const float* __restrict__ ba2,
                            float* __restrict__ wrow, float* __restrict__ wcol,
                            float* __restrict__ cconst) {
  __shared__ float sb[DD];
  int i = threadIdx.x;
  float s1 = 0.f, s2 = 0.f;
  for (int j = 0; j < DD; ++j) {
    float w2 = Wa2[j];
    s1 += Wa1[i * DD + j] * w2;
    s2 += Wa1[(DD + i) * DD + j] * w2;
  }
  wrow[i] = s1; wcol[i] = s2;
  sb[i] = ba1[i] * Wa2[i];
  __syncthreads();
  for (int st = 64; st > 0; st >>= 1) { if (i < st) sb[i] += sb[i + st]; __syncthreads(); }
  if (i == 0) *cconst = sb[0] + ba2[0];
}

// ---------------- per-iteration kernels --------------------------------------
// m = propH @ Wf + bf. Writes m in B-swizzled f16 form (for the merged GEMM) and
// computes u = m.wrow, v = m.wcol in-block via LDS (block owns full 16x128 rows).
__global__ void k_gemm_m(const _Float16* __restrict__ propH, const _Float16* __restrict__ WfBsw,
                         const float* __restrict__ bf, _Float16* __restrict__ mBsw,
                         const float* __restrict__ wrow, const float* __restrict__ wcol,
                         float* __restrict__ u, float* __restrict__ v) {
  __shared__ float sm[16][DD];
  __shared__ float smU[16][16], smV[16][16];
  int row0 = blockIdx.x * 16;
  int wave = threadIdx.x >> 5;
  int col0 = wave * 16;
  v8f acc = {};
#pragma unroll
  for (int k0 = 0; k0 < DD; k0 += 32) {
    v16h a = load_a_f16(propH, DD, row0, k0);
    v16h b = load_b_sw(WfBsw, (k0 >> 5) * NTILES + wave);
    acc = WMMA_F16(a, b, acc);
  }
  int lane = threadIdx.x & 31;
  int ncol = lane & 15;
  int mr   = ((lane >> 4) << 3);
  float bias = bf[col0 + ncol];

  // B-swizzled store of m: batch b, chunk c over seq dim, half selects dest lane hi/lo
  int b = row0 / NN;
  int i0 = row0 - b * NN;             // seq offset, multiple of 16
  int c  = i0 >> 5;                   // 32-row chunk
  int half = (i0 >> 4) & 1;           // which 16 of the 32 k-values
  int dlane = ncol + (half << 4);
  v8h hv;
#pragma unroll
  for (int r = 0; r < 8; ++r) {
    float mv = acc[r] + bias;
    sm[mr + r][col0 + ncol] = mv;
    hv[r] = (_Float16)mv;
  }
  *(v8h*)(mBsw + (((size_t)(b * (NN / 32) + c) * NTILES + wave) * 32 + dlane) * 16 + mr) = hv;

  __syncthreads();
  // u/v: 256 threads = 16 rows x 16 segments of 8 cols
  int tid = threadIdx.x;
  int rr = tid >> 4, seg = tid & 15;
  float su = 0.f, sv = 0.f;
#pragma unroll
  for (int q = 0; q < 8; ++q) {
    int col = seg * 8 + q;
    float mv = sm[rr][col];
    su += mv * wrow[col];
    sv += mv * wcol[col];
  }
  smU[rr][seg] = su; smV[rr][seg] = sv;
  __syncthreads();
  if (tid < 16) {
    float a1 = 0.f, a2 = 0.f;
#pragma unroll
    for (int q = 0; q < 16; ++q) { a1 += smU[tid][q]; a2 += smV[tid][q]; }
    u[row0 + tid] = a1; v[row0 + tid] = a2;
  }
}

// scores[b,i,j] = sigmoid(u[b,i] + v[b,j] + c) -> d_out slice + f16 copy (A operand)
__global__ void k_scores(const float* __restrict__ u, const float* __restrict__ v,
                         const float* __restrict__ cconst, float* __restrict__ out_sc,
                         _Float16* __restrict__ scoresH) {
  int idx = blockIdx.x * blockDim.x + threadIdx.x;   // < B*N*N (exact)
  int b = idx / (NN * NN), r = idx % (NN * NN);
  int i = r / NN, j = r % NN;
  float s = sigmoidf_(u[b * NN + i] + v[b * NN + j] + *cconst);
  out_sc[idx] = s;
  scoresH[idx] = (_Float16)s;
}

// merged[b] = scores[b] @ m[b]  (A: scoresH row-major; B: mBsw swizzled)
__global__ void k_merged(const _Float16* __restrict__ scoresH, const _Float16* __restrict__ mBsw,
                         float* __restrict__ out_merged, _Float16* __restrict__ mergedH) {
  int b = blockIdx.x / (NN / 16);
  int row0 = (blockIdx.x % (NN / 16)) * 16;
  int wave = threadIdx.x >> 5;
  const _Float16* A  = scoresH + (size_t)b * NN * NN;
  const _Float16* Bw = mBsw + (size_t)b * (NN / 32) * NTILES * FRAG;
  v8f acc = {};
#pragma unroll 4
  for (int k0 = 0; k0 < NN; k0 += 32) {
    v16h a = load_a_f16(A, NN, row0, k0);
    v16h bb = load_b_sw(Bw, (k0 >> 5) * NTILES + wave);
    acc = WMMA_F16(a, bb, acc);
  }
  int lane = threadIdx.x & 31;
  int col = wave * 16 + (lane & 15);
  int mr = ((lane >> 4) << 3);
#pragma unroll
  for (int r = 0; r < 8; ++r) {
    float mv = acc[r];
    size_t gi = (size_t)(b * NN + row0 + mr + r) * DD + col;
    out_merged[gi] = mv;
    mergedH[gi] = (_Float16)mv;
  }
}

// r = sig([merged,prop]@Wr+br), z = sig([merged,prop]@Wz+bz); store z (frag-flat)
// and (r*prop) as f16 row-major (A operand of h_hat GEMM). Shared A fragments.
__global__ void k_gates(const _Float16* __restrict__ mergedH, const _Float16* __restrict__ propH,
                        const float* __restrict__ pf,
                        const _Float16* __restrict__ WrBsw, const _Float16* __restrict__ WzBsw,
                        const float* __restrict__ br, const float* __restrict__ bz,
                        float* __restrict__ zf, _Float16* __restrict__ rpH) {
  int row0 = blockIdx.x * 16;
  int wave = threadIdx.x >> 5;
  v8f accR = {}, accZ = {};
#pragma unroll
  for (int k0 = 0; k0 < 2 * DD; k0 += 32) {
    v16h a = (k0 < DD) ? load_a_f16(mergedH, DD, row0, k0)
                       : load_a_f16(propH, DD, row0, k0 - DD);
    v16h bR = load_b_sw(WrBsw, (k0 >> 5) * NTILES + wave);
    v16h bZ = load_b_sw(WzBsw, (k0 >> 5) * NTILES + wave);
    accR = WMMA_F16(a, bR, accR);
    accZ = WMMA_F16(a, bZ, accZ);
  }
  int lane = threadIdx.x & 31;
  int col = wave * 16 + (lane & 15);
  int mr = ((lane >> 4) << 3);
  float biasR = br[col], biasZ = bz[col];
  const float* pfl = pf + ffidx(blockIdx.x, wave, lane, 0);
  float* zfl = zf + ffidx(blockIdx.x, wave, lane, 0);
#pragma unroll
  for (int r = 0; r < 8; ++r) {
    float rv = sigmoidf_(accR[r] + biasR);
    float zv = sigmoidf_(accZ[r] + biasZ);
    zfl[r] = zv;
    rpH[(size_t)(row0 + mr + r) * DD + col] = (_Float16)(rv * pfl[r]);
  }
}

// h_hat = tanh([merged, r*prop]@Wt+bt);  prop = (1-z)*prop + z*h_hat
__global__ void k_hhat(const _Float16* __restrict__ mergedH, const _Float16* __restrict__ rpH,
                       const _Float16* __restrict__ WtBsw, const float* __restrict__ bt,
                       const float* __restrict__ zf, float* __restrict__ pf,
                       _Float16* __restrict__ propH) {
  int row0 = blockIdx.x * 16;
  int wave = threadIdx.x >> 5;
  v8f acc = {};
#pragma unroll
  for (int k0 = 0; k0 < 2 * DD; k0 += 32) {
    v16h a = (k0 < DD) ? load_a_f16(mergedH, DD, row0, k0)
                       : load_a_f16(rpH, DD, row0, k0 - DD);
    v16h b = load_b_sw(WtBsw, (k0 >> 5) * NTILES + wave);
    acc = WMMA_F16(a, b, acc);
  }
  int lane = threadIdx.x & 31;
  int col = wave * 16 + (lane & 15);
  int mr = ((lane >> 4) << 3);
  float bias = bt[col];
  const float* zfl = zf + ffidx(blockIdx.x, wave, lane, 0);
  float* pfl = pf + ffidx(blockIdx.x, wave, lane, 0);
#pragma unroll
  for (int r = 0; r < 8; ++r) {
    float h = tanhf(acc[r] + bias);
    float z = zfl[r];
    float pn = (1.0f - z) * pfl[r] + z * h;
    pfl[r] = pn;
    propH[(size_t)(row0 + mr + r) * DD + col] = (_Float16)pn;
  }
}

// ---------------- launch -----------------------------------------------------
extern "C" void kernel_launch(void* const* d_in, const int* in_sizes, int n_in,
                              void* d_out, int out_size, void* d_ws, size_t ws_size,
                              hipStream_t stream) {
  const float* x   = (const float*)d_in[0];
  const float* Wf  = (const float*)d_in[1];
  const float* bf  = (const float*)d_in[2];
  const float* Wr  = (const float*)d_in[3];
  const float* br  = (const float*)d_in[4];
  const float* Wz  = (const float*)d_in[5];
  const float* bz  = (const float*)d_in[6];
  const float* Wt  = (const float*)d_in[7];
  const float* bt  = (const float*)d_in[8];
  const float* Wa1 = (const float*)d_in[9];
  const float* ba1 = (const float*)d_in[10];
  const float* Wa2 = (const float*)d_in[11];
  const float* ba2 = (const float*)d_in[12];

  float* out = (float*)d_out;
  float* out_scores = out;                                    // (1+T, B, N, N, 1)
  float* out_merged = out + (size_t)(1 + TT) * BB * NN * NN;  // (B, N, D)

  char* wp = (char*)d_ws;
  auto alloc = [&](size_t bytes) -> void* {
    void* p = (void*)wp; wp += (bytes + 255) & ~(size_t)255; return p;
  };
  float*    pf      = (float*)   alloc((size_t)ROWS * DD * 4);   // prop, fragment-flat
  float*    zf      = (float*)   alloc((size_t)ROWS * DD * 4);   // z, fragment-flat
  _Float16* propH   = (_Float16*)alloc((size_t)ROWS * DD * 2);   // row-major A operand
  _Float16* mergedH = (_Float16*)alloc((size_t)ROWS * DD * 2);
  _Float16* rpH     = (_Float16*)alloc((size_t)ROWS * DD * 2);
  _Float16* mBsw    = (_Float16*)alloc((size_t)BB * (NN / 32) * NTILES * FRAG * 2);
  _Float16* scoresH = (_Float16*)alloc((size_t)BB * NN * NN * 2);
  _Float16* WfBsw   = (_Float16*)alloc((size_t)DD * DD * 2);
  _Float16* WrBsw   = (_Float16*)alloc((size_t)2 * DD * DD * 2);
  _Float16* WzBsw   = (_Float16*)alloc((size_t)2 * DD * DD * 2);
  _Float16* WtBsw   = (_Float16*)alloc((size_t)2 * DD * DD * 2);
  float*    wrow    = (float*)   alloc(DD * 4);
  float*    wcol    = (float*)   alloc(DD * 4);
  float*    cconst  = (float*)   alloc(4);
  float*    u       = (float*)   alloc(ROWS * 4);
  float*    v       = (float*)   alloc(ROWS * 4);

  // ---- setup: swizzled f16 weights, folded attention vectors, prop = x ----
  k_swz_b<<<(DD * DD + 255) / 256, 256, 0, stream>>>(Wf, WfBsw, DD * DD);
  k_swz_b<<<(2 * DD * DD + 255) / 256, 256, 0, stream>>>(Wr, WrBsw, 2 * DD * DD);
  k_swz_b<<<(2 * DD * DD + 255) / 256, 256, 0, stream>>>(Wz, WzBsw, 2 * DD * DD);
  k_swz_b<<<(2 * DD * DD + 255) / 256, 256, 0, stream>>>(Wt, WtBsw, 2 * DD * DD);
  k_init_prop<<<(ROWS * DD + 255) / 256, 256, 0, stream>>>(x, pf, propH, ROWS * DD);
  k_prep_attn<<<1, 128, 0, stream>>>(Wa1, ba1, Wa2, ba2, wrow, wcol, cconst);

  // ---- 1+T recurrent iterations ----
  for (int t = 0; t <= TT; ++t) {
    k_gemm_m<<<ROWS / 16, 256, 0, stream>>>(propH, WfBsw, bf, mBsw, wrow, wcol, u, v);
    k_scores<<<(BB * NN * NN) / 256, 256, 0, stream>>>(
        u, v, cconst, out_scores + (size_t)t * BB * NN * NN, scoresH);
    k_merged<<<BB * (NN / 16), 256, 0, stream>>>(scoresH, mBsw, out_merged, mergedH);
    k_gates<<<ROWS / 16, 256, 0, stream>>>(mergedH, propH, pf, WrBsw, WzBsw, br, bz, zf, rpH);
    k_hhat<<<ROWS / 16, 256, 0, stream>>>(mergedH, rpH, WtBsw, bt, zf, pf, propH);
  }
}
